// SGCNet2_90580860272649
// MI455X (gfx1250) — matrix operand order repaired and checked by reading, not compiled, not run on hardware
//
#include <hip/hip_runtime.h>

#define FEAT 128
#define CLASSES 40
#define NPAD 48
#define NCHUNK 4              // FEAT / 32
#define NTILES 3              // NPAD / 16
#define NFRAG (NCHUNK * NTILES)

typedef __attribute__((ext_vector_type(16))) _Float16 v16h;
typedef __attribute__((ext_vector_type(8)))  float    v8f;

// ---------------- degree / norm ----------------

__global__ void k_init_deg(float* deg, int n) {
    int i = blockIdx.x * blockDim.x + threadIdx.x;
    if (i < n) deg[i] = 1.0f;  // self-loop contributes 1 to every degree
}

__global__ void k_accum_deg(const int* __restrict__ col, float* deg, int E) {
    int e = blockIdx.x * blockDim.x + threadIdx.x;
    if (e < E) atomicAdd(deg + col[e], 1.0f);
}

__global__ void k_rsqrt_deg(float* deg, int n) {
    int i = blockIdx.x * blockDim.x + threadIdx.x;
    if (i < n) deg[i] = rsqrtf(deg[i]);  // deg >= 1 always (self loop)
}

// ---------------- propagation ----------------

// out[i][:] = xin[i][:] * dinv[i]^2   (self-loop term; also zero-initializes out)
__global__ void k_seed(const float* __restrict__ xin, const float* __restrict__ dinv,
                       float* __restrict__ out, int n) {
    int idx = blockIdx.x * blockDim.x + threadIdx.x;      // float4 index
    int total = n * (FEAT / 4);
    if (idx < total) {
        int node = idx >> 5;                              // FEAT/4 == 32 float4 per node
        float d = dinv[node];
        float w = d * d;
        float4 v = ((const float4*)xin)[idx];
        v.x *= w; v.y *= w; v.z *= w; v.w *= w;
        ((float4*)out)[idx] = v;
    }
}

// out[col] += xin[row] * dinv[row]*dinv[col]; one wave32 per edge, float4 per lane
__global__ __launch_bounds__(256) void k_edge_prop(const float* __restrict__ xin,
                                                   const int* __restrict__ row,
                                                   const int* __restrict__ col,
                                                   const float* __restrict__ dinv,
                                                   float* __restrict__ out, int E) {
    int lane = threadIdx.x & 31;
    int wav  = threadIdx.x >> 5;
    int e = blockIdx.x * 8 + wav;
    if (e >= E) return;
    int r = row[e];
    int c = col[e];
    float w = dinv[r] * dinv[c];
    float4 v = ((const float4*)(xin + (size_t)r * FEAT))[lane];
    float* o = out + (size_t)c * FEAT + lane * 4;
    atomicAdd(o + 0, v.x * w);
    atomicAdd(o + 1, v.y * w);
    atomicAdd(o + 2, v.z * w);
    atomicAdd(o + 3, v.w * w);
}

// ---------------- B-fragment pre-pack ----------------
// Pack weight (f32 [128][40]) into f16 WMMA B-fragments laid out per-lane:
// bpack[((chunk*NTILES + t)*32 + lane)*16 + e]
//   lane: sel=lane>>4 (K half), n=lane&15 (column in tile)
//   element e -> B[k = chunk*32 + sel*8 + (e&7) + (e>>3)*16][col = t*16 + n]  (0 for col >= 40)
__global__ void k_pack_b(const float* __restrict__ weight, _Float16* __restrict__ bpack) {
    int idx = blockIdx.x * blockDim.x + threadIdx.x;   // [0, NFRAG*32*16)
    if (idx >= NFRAG * 32 * 16) return;
    int e    = idx & 15;
    int lane = (idx >> 4) & 31;
    int frag = idx >> 9;
    int t     = frag % NTILES;
    int chunk = frag / NTILES;
    int sel = lane >> 4;
    int nn  = lane & 15;
    int k   = chunk * 32 + sel * 8 + (e & 7) + (e >> 3) * 16;
    int c   = t * 16 + nn;
    float v = (c < CLASSES) ? weight[k * CLASSES + c] : 0.0f;
    bpack[idx] = (_Float16)v;
}

// ---------------- WMMA GEMM + log_softmax ----------------
// Each wave computes a 16x48 logits tile (N padded 40->48) with 12 back-to-back
// v_wmma_f32_16x16x32_f16 (B frags pre-packed, loaded as coalesced b128),
// then log_softmax over the real 40 classes with wave32 shuffle reductions.

__global__ __launch_bounds__(256) void k_gemm_lsm(const float* __restrict__ h,
                                                  const _Float16* __restrict__ bpack,
                                                  const float* __restrict__ bias,
                                                  float* __restrict__ out, int n) {
    __shared__ float slog[8][16][NPAD];   // 24 KB logits staging

    int wav  = threadIdx.x >> 5;
    int lane = threadIdx.x & 31;
    int mtiles = n / 16;
    int mtile = blockIdx.x * 8 + wav;
    if (mtile >= mtiles) return;          // wave-uniform: EXEC stays all-ones
    int mBase = mtile * 16;

    int rsel = lane >> 4;                 // lane half
    int rrow = lane & 15;
    int kb   = rsel * 8;
    const float* arow = h + (size_t)(mBase + rrow) * FEAT;

    // hoist all 12 pre-packed B fragments (each lane: one v16h = 32B, coalesced)
    const v16h* bp = (const v16h*)bpack;
    v16h bf[NFRAG];
#pragma unroll
    for (int f = 0; f < NFRAG; ++f) bf[f] = bp[f * 32 + lane];

    v8f acc0 = {}; v8f acc1 = {}; v8f acc2 = {};
#pragma unroll
    for (int chunk = 0; chunk < NCHUNK; ++chunk) {
        int kc = chunk * 32;
        // A fragment: 16x32 f16, ISA layout; two 32B-aligned float4 pairs + cvt
        const float4* p0 = (const float4*)(arow + kc + kb);
        const float4* p1 = (const float4*)(arow + kc + kb + 16);
        float4 f0 = p0[0], f1 = p0[1];
        float4 f2 = p1[0], f3 = p1[1];
        v16h a;
        a[0]  = (_Float16)f0.x; a[1]  = (_Float16)f0.y; a[2]  = (_Float16)f0.z; a[3]  = (_Float16)f0.w;
        a[4]  = (_Float16)f1.x; a[5]  = (_Float16)f1.y; a[6]  = (_Float16)f1.z; a[7]  = (_Float16)f1.w;
        a[8]  = (_Float16)f2.x; a[9]  = (_Float16)f2.y; a[10] = (_Float16)f2.z; a[11] = (_Float16)f2.w;
        a[12] = (_Float16)f3.x; a[13] = (_Float16)f3.y; a[14] = (_Float16)f3.z; a[15] = (_Float16)f3.w;

        acc0 = __builtin_amdgcn_wmma_f32_16x16x32_f16(false, a, false, bf[chunk * NTILES + 0],
                                                      (short)0, acc0, false, false);
        acc1 = __builtin_amdgcn_wmma_f32_16x16x32_f16(false, a, false, bf[chunk * NTILES + 1],
                                                      (short)0, acc1, false, false);
        acc2 = __builtin_amdgcn_wmma_f32_16x16x32_f16(false, a, false, bf[chunk * NTILES + 2],
                                                      (short)0, acc2, false, false);
    }

    // C layout: VGPR j holds row (j + rsel*8), col (ntile*16 + rrow). Stage logits (+bias) in LDS.
#pragma unroll
    for (int j = 0; j < 8; ++j) {
        int mrow = j + rsel * 8;
        slog[wav][mrow][rrow]      = acc0[j] + bias[rrow];
        slog[wav][mrow][16 + rrow] = acc1[j] + bias[16 + rrow];
        int c2 = 32 + rrow;
        slog[wav][mrow][c2]        = acc2[j] + ((c2 < CLASSES) ? bias[c2] : 0.0f);
    }

    // log_softmax over 40 classes per row (lane covers col=lane and, for lane<8, col=32+lane)
    for (int r = 0; r < 16; ++r) {
        float v1 = slog[wav][r][lane];
        bool has2 = lane < (CLASSES - 32);
        float v2 = has2 ? slog[wav][r][32 + lane] : -3.4e38f;
        float m = fmaxf(v1, v2);
#pragma unroll
        for (int d = 16; d >= 1; d >>= 1) m = fmaxf(m, __shfl_xor(m, d, 32));
        float s = __expf(v1 - m) + (has2 ? __expf(v2 - m) : 0.0f);
#pragma unroll
        for (int d = 16; d >= 1; d >>= 1) s += __shfl_xor(s, d, 32);
        float lse = m + __logf(s);
        float* orow = out + (size_t)(mBase + r) * CLASSES;
        orow[lane] = v1 - lse;
        if (has2) orow[32 + lane] = v2 - lse;
    }
}

// ---------------- launcher ----------------

extern "C" void kernel_launch(void* const* d_in, const int* in_sizes, int n_in,
                              void* d_out, int out_size, void* d_ws, size_t ws_size,
                              hipStream_t stream) {
    const float* x      = (const float*)d_in[0];
    const int*   ei     = (const int*)d_in[1];
    const float* weight = (const float*)d_in[2];
    const float* bias   = (const float*)d_in[3];
    float*       out    = (float*)d_out;

    int n = in_sizes[0] / FEAT;   // 100000 nodes
    int E = in_sizes[1] / 2;      // 1.6M edges
    const int* row = ei;          // edge_index[0] = source
    const int* col = ei + E;      // edge_index[1] = target

    char* ws = (char*)d_ws;
    float* deg = (float*)ws;                                     // n floats (becomes dinv)
    size_t degBytes = (((size_t)n * sizeof(float)) + 255) & ~(size_t)255;
    size_t hBytes   = (size_t)n * FEAT * sizeof(float);
    float* h1 = (float*)(ws + degBytes);
    float* h2 = (float*)(ws + degBytes + hBytes);
    _Float16* bpack = (_Float16*)(ws + degBytes + 2 * hBytes);   // 12 KB

    // normalization: deg = 1 + scatter(1) over col; dinv = rsqrt(deg)
    k_init_deg <<<(n + 255) / 256, 256, 0, stream>>>(deg, n);
    k_accum_deg<<<(E + 255) / 256, 256, 0, stream>>>(col, deg, E);
    k_rsqrt_deg<<<(n + 255) / 256, 256, 0, stream>>>(deg, n);

    // pre-pack WMMA B fragments from weight (overlaps with propagation)
    k_pack_b   <<<(NFRAG * 32 * 16 + 255) / 256, 256, 0, stream>>>(weight, bpack);

    int nv4 = n * (FEAT / 4);
    // hop 1: x -> h1
    k_seed     <<<(nv4 + 255) / 256, 256, 0, stream>>>(x, deg, h1, n);
    k_edge_prop<<<(E + 7) / 8, 256, 0, stream>>>(x, row, col, deg, h1, E);
    // hop 2: h1 -> h2
    k_seed     <<<(nv4 + 255) / 256, 256, 0, stream>>>(h1, deg, h2, n);
    k_edge_prop<<<(E + 7) / 8, 256, 0, stream>>>(h1, row, col, deg, h2, E);

    // WMMA GEMM + log_softmax: h2 @ W + b -> out
    int mtiles = n / 16;
    k_gemm_lsm <<<(mtiles + 7) / 8, 256, 0, stream>>>(h2, bpack, bias, out, n);
}
